// DeformationNetworkGcnHybrid_36335423324702
// MI455X (gfx1250) — matrix operand order repaired and checked by reading, not compiled, not run on hardware
//
#include <hip/hip_runtime.h>
#include <hip/hip_bf16.h>

// ---------------------------------------------------------------------------
// Problem constants (from the reference): B=4, V=16000, H=256
// ---------------------------------------------------------------------------
#define NV   64000   // B*V vertices
#define EE   192000  // 3*N edges
#define HH   256     // hidden width

typedef __attribute__((ext_vector_type(16))) __bf16 v16bf;
typedef __attribute__((ext_vector_type(8)))  float  v8f;

// ---- f32 <-> bf16 helpers (RNE) -------------------------------------------
__device__ __forceinline__ unsigned short f2bf(float f) {
  unsigned u = __float_as_uint(f);
  unsigned r = u + 0x7FFFu + ((u >> 16) & 1u);
  return (unsigned short)(r >> 16);
}
__device__ __forceinline__ float bf2f(unsigned short h) {
  return __uint_as_float(((unsigned)h) << 16);
}

// ---------------------------------------------------------------------------
// Weight conversion: f32 [R,C] row-major -> bf16 [R,Cp] (zero K-padding)
// ---------------------------------------------------------------------------
__global__ __launch_bounds__(256)
void cvt_weight_kernel(const float* __restrict__ src, unsigned short* __restrict__ dst,
                       int R, int C, int Cp) {
  int t = blockIdx.x * 256 + threadIdx.x;
  if (t >= R * Cp) return;
  int r = t / Cp, c = t - r * Cp;
  float v = (c < C) ? src[r * C + c] : 0.0f;
  dst[t] = f2bf(v);
}

// ---------------------------------------------------------------------------
// MLP1 layer 0: verts[N,3] @ W0^T[3,512] + b0, leaky, -> bf16 [N,512]
// (K=3: not worth WMMA, pure VALU, 0.2 GFLOP)
// ---------------------------------------------------------------------------
__global__ __launch_bounds__(256)
void mlp1_l0_kernel(const float* __restrict__ verts, const float* __restrict__ w,
                    const float* __restrict__ b, unsigned short* __restrict__ out) {
  int t = blockIdx.x * 256 + threadIdx.x;
  if (t >= NV * 512) return;
  int n = t >> 9, o = t & 511;
  float x = verts[n * 3 + 0] * w[o * 3 + 0]
          + verts[n * 3 + 1] * w[o * 3 + 1]
          + verts[n * 3 + 2] * w[o * 3 + 2] + b[o];
  x = (x >= 0.f) ? x : 0.01f * x;
  out[(size_t)n * 512 + o] = f2bf(x);
}

// ---------------------------------------------------------------------------
// Fill concat columns: x[:,256..258] = verts, x[:,259..287] = 0  (ld = 288)
// ---------------------------------------------------------------------------
__global__ __launch_bounds__(256)
void concat_fill_kernel(const float* __restrict__ verts, unsigned short* __restrict__ x) {
  int t = blockIdx.x * 256 + threadIdx.x;
  if (t >= NV * 32) return;
  int n = t >> 5, j = t & 31;
  float v = (j < 3) ? verts[n * 3 + j] : 0.0f;
  x[(size_t)n * 288 + 256 + j] = f2bf(v);
}

// ---------------------------------------------------------------------------
// Generic WMMA bf16 GEMM:  out[N, nout] = act( A[N,kp] @ W[nout,kp]^T + bias )
//   - wave computes a 16-row x (NJT*16)-col tile: 1 A-frag reused over NJT B-frags
//   - 8 waves per block -> 128 rows per block; grid.y covers nout/(16*NJT)
//   - NJT compile-time so acc[] lives in fixed even-aligned VGPR octets
//   - k-loop is software double-buffered: next k-step's fragments are issued
//     BEFORE the current step's NJT WMMAs, so loads overlap the matrix pipe and
//     the pre-WMMA wait need not drain the fresh loads.
//   - A fragment (16-bit A 16x32, ISA 7.12.2):
//       lane<16 : row M=lane,    K halves {0..7} and {16..23}
//       lane>=16: row M=lane-16, K halves {8..15} and {24..31}
//   - B fragment (32x16): lane<16 holds K {0..7} and {16..23} of column N=lane
//       (row-major weight row -> no transpose needed), lane>=16: K {8..15},{24..31}
//   - C/D: VGPR v -> row v (lanes 0-15) / row v+8 (lanes 16-31), col = lane&15
// ACT: 0=none, 2=leaky(0.01). OUTBF16 selects output type.
// ---------------------------------------------------------------------------
template <int ACT, bool OUTBF16, int NJT>
__global__ __launch_bounds__(256)
void gemm_wmma_kernel(const unsigned short* __restrict__ A,
                      const unsigned short* __restrict__ W,
                      const float* __restrict__ bias,
                      unsigned short* __restrict__ outB,
                      float* __restrict__ outF,
                      int nrows, int kp, int ldo) {
  const int lane = threadIdx.x & 31;
  const int wave = threadIdx.x >> 5;
  const int rowBase = blockIdx.x * 128 + wave * 16;
  const int colBase = blockIdx.y * (16 * NJT);
  if (rowBase >= nrows) return;                 // wave-uniform (EXEC stays all-1)

  const int arow  = rowBase + (lane & 15);
  const int akoff = (lane >> 4) * 8;            // A K-half select per ISA layout
  const int bkoff = (lane >> 4) * 8;            // B K-half select per ISA layout
  const unsigned short* aptr = A + (size_t)arow * kp + akoff;

  // Per-tile weight-row base pointers (wave-invariant stride kp).
  const unsigned short* wptr[NJT];
#pragma unroll
  for (int j = 0; j < NJT; ++j)
    wptr[j] = W + (size_t)(colBase + j * 16 + (lane & 15)) * kp + bkoff;

  v8f acc[NJT];
#pragma unroll
  for (int j = 0; j < NJT; ++j) acc[j] = (v8f){0, 0, 0, 0, 0, 0, 0, 0};

  union Frag { v16bf v; uint4 u[2]; };

  // ---- prologue: fragments for k0 = 0 ----
  Frag af, bf[NJT];
  af.u[0] = *reinterpret_cast<const uint4*>(aptr);
  af.u[1] = *reinterpret_cast<const uint4*>(aptr + 16);
#pragma unroll
  for (int j = 0; j < NJT; ++j) {
    bf[j].u[0] = *reinterpret_cast<const uint4*>(wptr[j]);
    bf[j].u[1] = *reinterpret_cast<const uint4*>(wptr[j] + 16);
  }

  // ---- pipelined main loop ----
  for (int k0 = 0; k0 < kp; k0 += 32) {
    const int kn = (k0 + 32 < kp) ? (k0 + 32) : k0;   // clamp last prefetch

    // issue next-step loads first (overlap with this step's WMMAs)
    Frag afn, bfn[NJT];
    afn.u[0] = *reinterpret_cast<const uint4*>(aptr + kn);
    afn.u[1] = *reinterpret_cast<const uint4*>(aptr + kn + 16);
#pragma unroll
    for (int j = 0; j < NJT; ++j) {
      bfn[j].u[0] = *reinterpret_cast<const uint4*>(wptr[j] + kn);
      bfn[j].u[1] = *reinterpret_cast<const uint4*>(wptr[j] + kn + 16);
    }

    // consume current fragments
#pragma unroll
    for (int j = 0; j < NJT; ++j) {
      acc[j] = __builtin_amdgcn_wmma_f32_16x16x32_bf16(
          false, af.v, false, bf[j].v, (short)0, acc[j], false, false);
    }

    // rotate
    af = afn;
#pragma unroll
    for (int j = 0; j < NJT; ++j) bf[j] = bfn[j];
  }

  // epilogue: bias + activation + store (f32 or bf16)
#pragma unroll
  for (int j = 0; j < NJT; ++j) {
    const int col = colBase + j * 16 + (lane & 15);
    const float bv = bias[col];
#pragma unroll
    for (int v = 0; v < 8; ++v) {
      const int row = rowBase + ((lane < 16) ? v : v + 8);
      if (row < nrows) {
        float x = acc[j][v] + bv;
        if (ACT == 2) x = (x >= 0.f) ? x : 0.01f * x;
        const size_t o = (size_t)row * ldo + col;
        if (OUTBF16) outB[o] = f2bf(x);
        else         outF[o] = x;
      }
    }
  }
}

// ---------------------------------------------------------------------------
// Edge scatter: agg[e0] += w1x[e1]; agg[e1] += w1x[e0]  (both directions)
// One thread per (edge, 4-col chunk): 2 x float4 loads, 8 f32 atomics -> L2.
// ---------------------------------------------------------------------------
__global__ __launch_bounds__(256)
void scatter_edges_kernel(const int* __restrict__ edges,
                          const float* __restrict__ w1x,
                          float* __restrict__ agg) {
  int t = blockIdx.x * 256 + threadIdx.x;
  if (t >= EE * 64) return;
  int e = t >> 6;
  int c = (t & 63) << 2;
  int a = edges[2 * e + 0];
  int b = edges[2 * e + 1];
  float4 vb = *reinterpret_cast<const float4*>(w1x + (size_t)b * HH + c);
  float4 va = *reinterpret_cast<const float4*>(w1x + (size_t)a * HH + c);
  float* pa = agg + (size_t)a * HH + c;
  float* pb = agg + (size_t)b * HH + c;
  atomicAdd(pa + 0, vb.x); atomicAdd(pa + 1, vb.y);
  atomicAdd(pa + 2, vb.z); atomicAdd(pa + 3, vb.w);
  atomicAdd(pb + 0, va.x); atomicAdd(pb + 1, va.y);
  atomicAdd(pb + 2, va.z); atomicAdd(pb + 3, va.w);
}

// ---------------------------------------------------------------------------
// relu + convert f32 [N,256] -> bf16 [N,256]
// ---------------------------------------------------------------------------
__global__ __launch_bounds__(256)
void relu_cvt_kernel(const float* __restrict__ src, unsigned short* __restrict__ dst) {
  size_t t = (size_t)blockIdx.x * 256 + threadIdx.x;
  if (t >= (size_t)NV * HH) return;
  float x = src[t];
  dst[t] = f2bf(x > 0.f ? x : 0.f);
}

// ---------------------------------------------------------------------------
// MLP3 final layer: act32 (bf16, ld=32) @ W3^T[32,3] + b3 -> f32 out [N,3]
// ---------------------------------------------------------------------------
__global__ __launch_bounds__(256)
void mlp3_final_kernel(const unsigned short* __restrict__ act,
                       const float* __restrict__ w, const float* __restrict__ b,
                       float* __restrict__ out) {
  int n = blockIdx.x * 256 + threadIdx.x;
  if (n >= NV) return;
  float h[32];
#pragma unroll
  for (int i = 0; i < 32; ++i) h[i] = bf2f(act[(size_t)n * 32 + i]);
#pragma unroll
  for (int k = 0; k < 3; ++k) {
    float s = b[k];
#pragma unroll
    for (int i = 0; i < 32; ++i) s += h[i] * w[k * 32 + i];
    out[n * 3 + k] = s;
  }
}

// ---------------------------------------------------------------------------
// Host orchestration
// ---------------------------------------------------------------------------
extern "C" void kernel_launch(void* const* d_in, const int* in_sizes, int n_in,
                              void* d_out, int out_size, void* d_ws, size_t ws_size,
                              hipStream_t stream) {
  (void)in_sizes; (void)n_in; (void)out_size; (void)ws_size;

  const float* verts = (const float*)d_in[0];
  const int*   edges = (const int*)  d_in[1];
  const float* m1_w0 = (const float*)d_in[2];  const float* m1_b0 = (const float*)d_in[3];
  const float* m1_w1 = (const float*)d_in[4];  const float* m1_b1 = (const float*)d_in[5];
  const float* m1_w2 = (const float*)d_in[6];  const float* m1_b2 = (const float*)d_in[7];
  const float* m1_w3 = (const float*)d_in[8];  const float* m1_b3 = (const float*)d_in[9];
  const float* g0_w0 = (const float*)d_in[10]; const float* g0_b0 = (const float*)d_in[11];
  const float* g0_w1 = (const float*)d_in[12]; const float* g0_b1 = (const float*)d_in[13];
  const float* g_w0  = (const float*)d_in[14]; const float* g_b0  = (const float*)d_in[15];
  const float* g_w1  = (const float*)d_in[16]; const float* g_b1  = (const float*)d_in[17];
  const float* m3_w0 = (const float*)d_in[18]; const float* m3_b0 = (const float*)d_in[19];
  const float* m3_w1 = (const float*)d_in[20]; const float* m3_b1 = (const float*)d_in[21];
  const float* m3_w2 = (const float*)d_in[22]; const float* m3_b2 = (const float*)d_in[23];
  const float* m3_w3 = (const float*)d_in[24]; const float* m3_b3 = (const float*)d_in[25];
  float* out = (float*)d_out;

  // ---- workspace carve-out (256B aligned) ----
  char* ws = (char*)d_ws;
  size_t off = 0;
  auto take = [&](size_t bytes) -> char* {
    char* p = ws + off;
    off = (off + bytes + 255) & ~(size_t)255;
    return p;
  };
  unsigned short* Wm1w1 = (unsigned short*)take((size_t)512 * 512 * 2);
  unsigned short* Wm1w2 = (unsigned short*)take((size_t)256 * 512 * 2);
  unsigned short* Wm1w3 = (unsigned short*)take((size_t)256 * 256 * 2);
  unsigned short* Wg0w0 = (unsigned short*)take((size_t)256 * 288 * 2);
  unsigned short* Wg0w1 = (unsigned short*)take((size_t)256 * 288 * 2);
  unsigned short* Wgw0  = (unsigned short*)take((size_t)9 * 256 * 256 * 2);
  unsigned short* Wgw1  = (unsigned short*)take((size_t)9 * 256 * 256 * 2);
  unsigned short* Wm3w0 = (unsigned short*)take((size_t)128 * 256 * 2);
  unsigned short* Wm3w1 = (unsigned short*)take((size_t)64 * 128 * 2);
  unsigned short* Wm3w2 = (unsigned short*)take((size_t)32 * 64 * 2);
  unsigned short* PING  = (unsigned short*)take((size_t)NV * 512 * 2);
  unsigned short* PONG  = (unsigned short*)take((size_t)NV * 512 * 2);
  float* F32A = (float*)take((size_t)NV * HH * 4);   // w1x
  float* F32B = (float*)take((size_t)NV * HH * 4);   // pre-activation / agg target

  auto cvt = [&](const float* s, unsigned short* d, int R, int C, int Cp) {
    int n = R * Cp;
    cvt_weight_kernel<<<(n + 255) / 256, 256, 0, stream>>>(s, d, R, C, Cp);
  };
  // bf16 weight conversion (K padded to multiple of 32)
  cvt(m1_w1, Wm1w1, 512, 512, 512);
  cvt(m1_w2, Wm1w2, 256, 512, 512);
  cvt(m1_w3, Wm1w3, 256, 256, 256);
  cvt(g0_w0, Wg0w0, 256, 259, 288);
  cvt(g0_w1, Wg0w1, 256, 259, 288);
  cvt(g_w0,  Wgw0,  9 * 256, 256, 256);
  cvt(g_w1,  Wgw1,  9 * 256, 256, 256);
  cvt(m3_w0, Wm3w0, 128, 256, 256);
  cvt(m3_w1, Wm3w1, 64, 128, 128);
  cvt(m3_w2, Wm3w2, 32, 64, 64);

  const int gx = (NV + 127) / 128;   // 500 row-blocks

  // leaky + bf16 output, 64-col tiles per block
  auto gemm_leaky_bf = [&](const unsigned short* A, const unsigned short* W,
                           const float* bias, int nout, int kp, int ldo,
                           unsigned short* dst) {
    dim3 g(gx, nout / 64);
    gemm_wmma_kernel<2, true, 4><<<g, 256, 0, stream>>>(A, W, bias, dst, nullptr,
                                                        NV, kp, ldo);
  };
  // no activation + f32 output (graph-conv partials)
  auto gemm_f32 = [&](const unsigned short* A, const unsigned short* W,
                      const float* bias, int nout, int kp, float* dst) {
    dim3 g(gx, nout / 64);
    gemm_wmma_kernel<0, false, 4><<<g, 256, 0, stream>>>(A, W, bias, nullptr, dst,
                                                         NV, kp, nout);
  };

  // ---- MLP1: 3 -> 512 -> 512 -> 256 -> 256 (leaky) ----
  mlp1_l0_kernel<<<(NV * 512 + 255) / 256, 256, 0, stream>>>(verts, m1_w0, m1_b0, PING);
  gemm_leaky_bf(PING, Wm1w1, m1_b1, 512, 512, 512, PONG);
  gemm_leaky_bf(PONG, Wm1w2, m1_b2, 256, 512, 256, PING);
  gemm_leaky_bf(PING, Wm1w3, m1_b3, 256, 256, 288, PONG);   // into concat buffer cols 0..255
  concat_fill_kernel<<<(NV * 32 + 255) / 256, 256, 0, stream>>>(verts, PONG);

  // ---- GraphConv stack: g0 (K=288) then 9 x (K=256) ----
  unsigned short* cur = PONG;
  unsigned short* nxt = PING;
  const int scatter_blocks = (EE * 64 + 255) / 256;
  const int ew_blocks = (int)(((size_t)NV * HH + 255) / 256);

  // conv 0
  gemm_f32(cur, Wg0w1, g0_b1, 256, 288, F32A);              // w1x
  gemm_f32(cur, Wg0w0, g0_b0, 256, 288, F32B);              // x@w0^T + b0
  scatter_edges_kernel<<<scatter_blocks, 256, 0, stream>>>(edges, F32A, F32B);
  relu_cvt_kernel<<<ew_blocks, 256, 0, stream>>>(F32B, nxt);
  { unsigned short* t = cur; cur = nxt; nxt = t; }

  // convs 1..9
  for (int i = 0; i < 9; ++i) {
    const unsigned short* w1 = Wgw1 + (size_t)i * 256 * 256;
    const unsigned short* w0 = Wgw0 + (size_t)i * 256 * 256;
    gemm_f32(cur, w1, g_b1 + i * 256, 256, 256, F32A);
    gemm_f32(cur, w0, g_b0 + i * 256, 256, 256, F32B);
    scatter_edges_kernel<<<scatter_blocks, 256, 0, stream>>>(edges, F32A, F32B);
    relu_cvt_kernel<<<ew_blocks, 256, 0, stream>>>(F32B, nxt);
    unsigned short* t = cur; cur = nxt; nxt = t;
  }

  // ---- MLP3: 256 -> 128 -> 64 -> 32 (leaky) -> 3 ----
  gemm_leaky_bf(cur, Wm3w0, m3_b0, 128, 256, 128, nxt);
  { unsigned short* t = cur; cur = nxt; nxt = t; }
  gemm_leaky_bf(cur, Wm3w1, m3_b1, 64, 128, 64, nxt);
  { unsigned short* t = cur; cur = nxt; nxt = t; }
  // nout = 32: one 32-col tile per block (NJT = 2)
  {
    dim3 g(gx, 1);
    gemm_wmma_kernel<2, true, 2><<<g, 256, 0, stream>>>(cur, Wm3w2, m3_b2, nxt, nullptr,
                                                        NV, 64, 32);
  }
  { unsigned short* t = cur; cur = nxt; nxt = t; }
  mlp3_final_kernel<<<(NV + 255) / 256, 256, 0, stream>>>(cur, m3_w3, m3_b3, out);
}
